// AttentionBlock_58566174048334
// MI455X (gfx1250) — matrix operand order repaired and checked by reading, not compiled, not run on hardware
//
#include <hip/hip_runtime.h>
#include <hip/hip_bf16.h>

#define B_ 8
#define N_ 1024
#define DIM_ 768
#define HEADS_ 12
#define HD_ 64
#define INNER_ 768
#define SCALE_ 0.125f

typedef __attribute__((ext_vector_type(16))) __bf16 v16bf;
typedef __attribute__((ext_vector_type(8)))  float  v8f;
typedef __attribute__((ext_vector_type(8)))  unsigned short v8u;
typedef __attribute__((ext_vector_type(16))) unsigned short v16u;
typedef unsigned short u16;

// native f32 -> bf16 (ISel picks v_cvt_*bf16_f32)
static __device__ __forceinline__ u16 f2bf(float f) {
  return __builtin_bit_cast(u16, (__bf16)f);
}
static __device__ __forceinline__ unsigned pk2bf(float a, float b) {
  unsigned short lo = __builtin_bit_cast(unsigned short, (__bf16)a);
  unsigned short hi = __builtin_bit_cast(unsigned short, (__bf16)b);
  return (unsigned)lo | ((unsigned)hi << 16);
}

static __device__ __forceinline__ v8f v8f_zero() {
  v8f z = {0.f, 0.f, 0.f, 0.f, 0.f, 0.f, 0.f, 0.f};
  return z;
}

// ---- async global->LDS copy (CDNA5 GLOBAL_LOAD_ASYNC_TO_LDS_B128) ----------
#if __has_builtin(__builtin_amdgcn_global_load_async_to_lds_b128)
#define HAVE_ASYNC_LDS 1
#else
#define HAVE_ASYNC_LDS 0
#endif

// exact parameter type per hipcc diagnostic: int __vector_size__(16), AS1/AS3
typedef int vsi4 __attribute__((__vector_size__(4 * sizeof(int))));
typedef vsi4 __attribute__((address_space(1)))* gp_v4i;
typedef vsi4 __attribute__((address_space(3)))* lp_v4i;

static __device__ __forceinline__ void async_copy16(const u16* g, u16* l) {
#if HAVE_ASYNC_LDS
  __builtin_amdgcn_global_load_async_to_lds_b128(
      (gp_v4i)(void*)g, (lp_v4i)(void*)l, 0, 0);
#else
  *(v8u*)l = *(const v8u*)g;
#endif
}
static __device__ __forceinline__ void async_wait() {
#if HAVE_ASYNC_LDS
#if __has_builtin(__builtin_amdgcn_s_wait_asynccnt)
  __builtin_amdgcn_s_wait_asynccnt(0);
#else
  asm volatile("s_wait_asynccnt 0x0" ::: "memory");
#endif
#endif
}

// A-matrix fragment (16x32 bf16), per CDNA5 ISA layout:
// lanes 0-15: row m=lane,     K = {0..7, 16..23}
// lanes 16-31: row m=lane-16, K = {8..15, 24..31}
static __device__ __forceinline__ v16bf load_frag_a(const u16* base, int stride, int lane) {
  int m  = lane & 15;
  int kb = (lane >> 4) << 3;          // 0 or 8
  const u16* p = base + m * stride + kb;
  v8u lo = *(const v8u*)(p);
  v8u hi = *(const v8u*)(p + 16);
  v16u r = __builtin_shufflevector(lo, hi, 0,1,2,3,4,5,6,7,8,9,10,11,12,13,14,15);
  return __builtin_bit_cast(v16bf, r);
}

// B-matrix fragment (32x16 bf16) read from [n][k]-major storage:
// lanes 0-15: col n=lane, K = {0..7, 16..23} ; lanes 16-31: col n=lane-16, K = {8..15, 24..31}
static __device__ __forceinline__ v16bf load_frag_b(const u16* base, int stride, int lane) {
  int n  = lane & 15;
  int kb = (lane >> 4) << 3;          // 0 or 8
  const u16* p = base + n * stride + kb;
  v8u lo = *(const v8u*)(p);
  v8u hi = *(const v8u*)(p + 16);
  v16u r = __builtin_shufflevector(lo, hi, 0,1,2,3,4,5,6,7,8,9,10,11,12,13,14,15);
  return __builtin_bit_cast(v16bf, r);
}

static __device__ __forceinline__ v8f wmma_bf16(v16bf a, v16bf b, v8f c) {
  return __builtin_amdgcn_wmma_f32_16x16x32_bf16(false, a, false, b, (short)0, c, false, false);
}

// ---------------------------------------------------------------------------
// Stage 1: qkv = x @ w_qkv, scattered to head-major bf16 Q, K and V^T.
// Tile 256(M) x 64(N), K-step 32; 8 waves, wave owns 32x64 (8 WMMAs/step).
// ---------------------------------------------------------------------------
__global__ __launch_bounds__(256) void qkv_kernel(const float* __restrict__ x,
                                                  const float* __restrict__ w,
                                                  u16* __restrict__ qh,
                                                  u16* __restrict__ kh,
                                                  u16* __restrict__ vt) {
  __shared__ u16 As[256 * 40];   // stride 40 (=80B) kills bank conflicts
  __shared__ u16 Bs[64 * 40];
  const int R0 = blockIdx.x * 256;
  const int C0 = blockIdx.y * 64;
  const int t = threadIdx.x;
  const int wave = t >> 5, lane = t & 31;

  v8f acc[2][4];
#pragma unroll
  for (int m = 0; m < 2; ++m)
#pragma unroll
    for (int i = 0; i < 4; ++i) acc[m][i] = v8f_zero();

  // incremental staging pointers (constant-add per iteration)
  const float* xp = x + (size_t)(R0 + (t >> 3)) * DIM_ + (t & 7) * 4;
  u16*         ap = As + (t >> 3) * 40 + (t & 7) * 4;
  const float* wp = w + (size_t)(t >> 4) * (3 * INNER_) + C0 + (t & 15) * 4;
  u16*         bp = Bs + ((t & 15) * 4) * 40 + (t >> 4);

  for (int kt = 0; kt < DIM_ / 32; ++kt) {
    // A tile: 256x32 f32 -> bf16, packed u32 LDS stores
    const float* xq = xp;
    u16* aq = ap;
#pragma unroll
    for (int i = 0; i < 8; ++i) {
      float4 f = *(const float4*)xq;
      unsigned* d = (unsigned*)aq;
      d[0] = pk2bf(f.x, f.y);
      d[1] = pk2bf(f.z, f.w);
      xq += (size_t)32 * DIM_;
      aq += 32 * 40;
    }
    // B tile: 32x64 f32 -> bf16, stored transposed [n][k]
    const float* wq = wp;
    u16* bq = bp;
#pragma unroll
    for (int i = 0; i < 2; ++i) {
      float4 f = *(const float4*)wq;
      bq[0 * 40] = f2bf(f.x);
      bq[1 * 40] = f2bf(f.y);
      bq[2 * 40] = f2bf(f.z);
      bq[3 * 40] = f2bf(f.w);
      wq += (size_t)16 * (3 * INNER_);
      bq += 16;
    }
    __syncthreads();
    v16bf a0 = load_frag_a(As + (wave * 32 + 0) * 40, 40, lane);
    v16bf a1 = load_frag_a(As + (wave * 32 + 16) * 40, 40, lane);
#pragma unroll
    for (int cc = 0; cc < 4; ++cc) {
      v16bf b = load_frag_b(Bs + cc * 16 * 40, 40, lane);
      acc[0][cc] = wmma_bf16(a0, b, acc[0][cc]);
      acc[1][cc] = wmma_bf16(a1, b, acc[1][cc]);
    }
    __syncthreads();
    xp += 32;
    wp += (size_t)32 * (3 * INNER_);
  }

  // Epilogue: sel is block-uniform; 8-row runs are contiguous in n.
  const int sel = C0 / INNER_;
  const int cbase = C0 - sel * INNER_;
  const int nl = lane & 15, rg = (lane >> 4) << 3;
#pragma unroll
  for (int mm = 0; mm < 2; ++mm) {
    const int rb = R0 + wave * 32 + mm * 16 + rg;   // multiple of 8
    const int bb = rb >> 10;
    const int nb = rb & 1023;
#pragma unroll
    for (int cc = 0; cc < 4; ++cc) {
      const int inner = cbase + cc * 16 + nl;
      const int hh = inner >> 6, dd = inner & 63;
      const int bh = bb * HEADS_ + hh;
      if (sel < 2) {
        u16* dst = (sel == 0 ? qh : kh) + (size_t)(bh * N_ + nb) * HD_ + dd;
#pragma unroll
        for (int r = 0; r < 8; ++r) dst[r * HD_] = f2bf(acc[mm][cc][r]);
      } else {
        unsigned* dst = (unsigned*)(vt + (size_t)(bh * HD_ + dd) * N_ + nb);
#pragma unroll
        for (int r = 0; r < 4; ++r)
          dst[r] = pk2bf(acc[mm][cc][2 * r], acc[mm][cc][2 * r + 1]);
      }
    }
  }
}

// ---------------------------------------------------------------------------
// Stage 2: fused attention with softmax over the HEADS axis.
// One workgroup = (batch b, 16-query tile); 12 waves = 12 heads.
// ---------------------------------------------------------------------------
__global__ __launch_bounds__(384) void attn_kernel(const u16* __restrict__ qh,
                                                   const u16* __restrict__ kh,
                                                   const u16* __restrict__ vt,
                                                   u16* __restrict__ ao) {
  __shared__ float E[2][HEADS_][16 * 32];   // exp(scores), double-buffered
  __shared__ float Sr[2][16 * 32];          // 1 / sum_h exp(scores)
  const int t = threadIdx.x;
  const int h = t >> 5, lane = t & 31;
  const int b = blockIdx.x >> 6;                 // 64 query tiles / batch
  const int n0 = (blockIdx.x & 63) * 16;
  const int bh = b * HEADS_ + h;
  const int nl = lane & 15;
  const int rg = (lane >> 4) << 3;               // 0 or 8

  const u16* qbase = qh + (size_t)(bh * N_ + n0) * HD_;
  v16bf q0 = load_frag_a(qbase, HD_, lane);        // d = 0..31
  v16bf q1 = load_frag_a(qbase + 32, HD_, lane);   // d = 32..63

  const u16* kbase = kh + (size_t)bh * N_ * HD_;
  const u16* vbase = vt + (size_t)bh * HD_ * N_;

  v8f o[4];
#pragma unroll
  for (int i = 0; i < 4; ++i) o[i] = v8f_zero();

  for (int m0 = 0; m0 < N_; m0 += 32) {
    const int pb = (m0 >> 5) & 1;
    if (m0 + 32 < N_) {       // prefetch next K/V m-tile
      __builtin_prefetch(kbase + (size_t)(m0 + 32) * HD_, 0, 1);
      __builtin_prefetch(vbase + m0 + 32, 0, 1);
    }
    // scores: 16 queries x 32 keys for this head (K^T via B-fragments)
    v8f s[2];
#pragma unroll
    for (int cc = 0; cc < 2; ++cc) {
      const u16* kb = kbase + (size_t)(m0 + cc * 16) * HD_;
      v8f ss = v8f_zero();
      ss = wmma_bf16(q0, load_frag_b(kb, HD_, lane), ss);
      ss = wmma_bf16(q1, load_frag_b(kb + 32, HD_, lane), ss);
      s[cc] = ss;
    }
    // exp and publish (softmax couples the 12 heads at each (n,m))
#pragma unroll
    for (int cc = 0; cc < 2; ++cc)
#pragma unroll
      for (int r = 0; r < 8; ++r)
        E[pb][h][(rg + r) * 32 + cc * 16 + nl] = __expf(s[cc][r] * SCALE_);
    __syncthreads();
    for (int i = t; i < 512; i += 384) {
      float ssum = 0.f;
#pragma unroll
      for (int hh = 0; hh < HEADS_; ++hh) ssum += E[pb][hh][i];
      Sr[pb][i] = __builtin_amdgcn_rcpf(ssum);
    }
    __syncthreads();
    // normalized P fragment in A-layout (16 rows x 32 K=m), bf16
    v16u pu;
    {
      const float* Eh = &E[pb][h][nl * 32];
      const float* Sv = &Sr[pb][nl * 32];
#pragma unroll
      for (int j = 0; j < 8; ++j) { int m = rg + j;      pu[j]     = f2bf(Eh[m] * Sv[m]); }
#pragma unroll
      for (int j = 0; j < 8; ++j) { int m = rg + 16 + j; pu[8 + j] = f2bf(Eh[m] * Sv[m]); }
    }
    v16bf pa = __builtin_bit_cast(v16bf, pu);
    // out += P @ V  (V^T so contraction dim m is contiguous)
#pragma unroll
    for (int dc = 0; dc < 4; ++dc)
      o[dc] = wmma_bf16(pa, load_frag_b(vbase + (size_t)(dc * 16) * N_ + m0, N_, lane), o[dc]);
    // ping-pong E/Sr: two barriers/iter give write->read separation
  }

  // merge heads: contiguous 8-row runs, constant-offset stores
#pragma unroll
  for (int dc = 0; dc < 4; ++dc) {
    u16* dst = ao + (size_t)(b * N_ + n0 + rg) * INNER_ + h * HD_ + dc * 16 + nl;
#pragma unroll
    for (int r = 0; r < 8; ++r) dst[r * INNER_] = f2bf(o[dc][r]);
  }
}

// ---------------------------------------------------------------------------
// Stage 3: out = attn @ w_out + b_out  (fp32), tile 256x64,
// A tile staged with async global->LDS copies (ASYNCcnt).
// ---------------------------------------------------------------------------
__global__ __launch_bounds__(256) void out_kernel(const u16* __restrict__ attn,
                                                  const float* __restrict__ w,
                                                  const float* __restrict__ bias,
                                                  float* __restrict__ out) {
  __shared__ u16 As[256 * 40];
  __shared__ u16 Bs[64 * 40];
  const int R0 = blockIdx.x * 256;
  const int C0 = blockIdx.y * 64;
  const int t = threadIdx.x;
  const int wave = t >> 5, lane = t & 31;

  v8f acc[2][4];
#pragma unroll
  for (int m = 0; m < 2; ++m)
#pragma unroll
    for (int i = 0; i < 4; ++i) acc[m][i] = v8f_zero();

  const u16*   app = attn + (size_t)(R0 + (t >> 2)) * INNER_ + (t & 3) * 8;
  u16*         alp = As + (t >> 2) * 40 + (t & 3) * 8;
  const float* wp  = w + (size_t)(t >> 4) * DIM_ + C0 + (t & 15) * 4;
  u16*         bp  = Bs + ((t & 15) * 4) * 40 + (t >> 4);

  for (int kt = 0; kt < INNER_ / 32; ++kt) {
    // A tile already bf16: async copy 16B/lane straight into LDS
#pragma unroll
    for (int i = 0; i < 4; ++i)
      async_copy16(app + (size_t)(i * 64) * INNER_, alp + i * 64 * 40);
    // B tile: 32x64 f32 -> bf16, transposed [n][k]
    const float* wq = wp;
    u16* bq = bp;
#pragma unroll
    for (int i = 0; i < 2; ++i) {
      float4 f = *(const float4*)wq;
      bq[0 * 40] = f2bf(f.x);
      bq[1 * 40] = f2bf(f.y);
      bq[2 * 40] = f2bf(f.z);
      bq[3 * 40] = f2bf(f.w);
      wq += (size_t)16 * DIM_;
      bq += 16;
    }
    async_wait();
    __syncthreads();
    v16bf a0 = load_frag_a(As + (wave * 32 + 0) * 40, 40, lane);
    v16bf a1 = load_frag_a(As + (wave * 32 + 16) * 40, 40, lane);
#pragma unroll
    for (int cc = 0; cc < 4; ++cc) {
      v16bf bfrag = load_frag_b(Bs + cc * 16 * 40, 40, lane);
      acc[0][cc] = wmma_bf16(a0, bfrag, acc[0][cc]);
      acc[1][cc] = wmma_bf16(a1, bfrag, acc[1][cc]);
    }
    __syncthreads();
    app += 32;
    wp += (size_t)32 * DIM_;
  }

  const int nl = lane & 15, rg = (lane >> 4) << 3;
#pragma unroll
  for (int mm = 0; mm < 2; ++mm) {
    const int rb = R0 + wave * 32 + mm * 16 + rg;
#pragma unroll
    for (int cc = 0; cc < 4; ++cc) {
      const int col = C0 + cc * 16 + nl;
      const float bb = bias[col];
      float* dst = out + (size_t)rb * DIM_ + col;
#pragma unroll
      for (int r = 0; r < 8; ++r) dst[r * DIM_] = acc[mm][cc][r] + bb;
    }
  }
}

// ---------------------------------------------------------------------------
extern "C" void kernel_launch(void* const* d_in, const int* in_sizes, int n_in,
                              void* d_out, int out_size, void* d_ws, size_t ws_size,
                              hipStream_t stream) {
  (void)in_sizes; (void)n_in; (void)out_size;
  const float* x     = (const float*)d_in[0];
  const float* w_qkv = (const float*)d_in[1];
  const float* w_out = (const float*)d_in[2];
  const float* b_out = (const float*)d_in[3];
  float* out = (float*)d_out;

  const size_t TEN = (size_t)B_ * HEADS_ * N_ * HD_;  // 6,291,456 elements
  u16* qh = (u16*)d_ws;        // Q  head-major [B][H][N][64]       bf16
  u16* kh = qh + TEN;          // K  head-major [B][H][N][64]       bf16
  u16* vt = kh + TEN;          // V^T head-major [B][H][64][N]      bf16
  u16* ao = vt + TEN;          // attn output   [B*N][INNER]        bf16
  if (ws_size < 4 * TEN * sizeof(u16)) return;        // ~50 MB needed

  qkv_kernel<<<dim3(32, 36), 256, 0, stream>>>(x, w_qkv, qh, kh, vt);
  attn_kernel<<<dim3(B_ * (N_ / 16)), 384, 0, stream>>>(qh, kh, vt, ao);
  out_kernel<<<dim3(32, 12), 256, 0, stream>>>(ao, w_out, b_out, out);
}
